// LSTM_32091995636113
// MI455X (gfx1250) — matrix operand order repaired and checked by reading, not compiled, not run on hardware
//
#include <hip/hip_runtime.h>
#include <hip/hip_bf16.h>

typedef __attribute__((ext_vector_type(16))) _Float16 v16h;
typedef __attribute__((ext_vector_type(8)))  _Float16 v8h;
typedef __attribute__((ext_vector_type(8)))  float    v8f;

#define TSTEPS 512
#define HID 32

__device__ __forceinline__ v8f wmma_f16(v16h a, v16h b, v8f c) {
    // D = A(16x32) * B(32x16) + C ; emits v_wmma_f32_16x16x32_f16
    return __builtin_amdgcn_wmma_f32_16x16x32_f16(false, a, false, b, (short)0, c, false, false);
}

#if __has_builtin(__builtin_amdgcn_tanhf)
__device__ __forceinline__ float tanh_hw(float v) {
    return __builtin_amdgcn_tanhf(v);          // gfx1250 v_tanh_f32 (1 trans op)
}
#else
__device__ __forceinline__ float tanh_hw(float v) {
    float e = __expf(-2.0f * v);
    return (1.0f - e) * __builtin_amdgcn_rcpf(1.0f + e);
}
#endif
__device__ __forceinline__ float sigm(float v) {
    // sigmoid(x) = 0.5 + 0.5*tanh(x/2): mul + tanh + fma
    return fmaf(tanh_hw(0.5f * v), 0.5f, 0.5f);
}

// One wave (wave32) owns 16 batch rows. Orientation: A = weights (16 gates x K=32),
// B = state (K=32 hidden x 16 batch), D/C: lane = batch col, VGPR = gate row.
__global__ __launch_bounds__(256)
void lstm_wmma_kernel(const float* __restrict__ x,
                      const float* __restrict__ W_ih0,
                      const float* __restrict__ W_hh0,
                      const float* __restrict__ b_ih0,
                      const float* __restrict__ b_hh0,
                      const float* __restrict__ W_ih1,
                      const float* __restrict__ W_hh1,
                      const float* __restrict__ b_ih1,
                      const float* __restrict__ b_hh1,
                      const float* __restrict__ W_fc,
                      const float* __restrict__ b_fc,
                      float* __restrict__ out)
{
    // per-wave transpose staging: 16 batch rows x 32 hidden f16, row stride 40 halfs
    // (80B -> 16B aligned, bank-conflict-free for both the b128 stores and loads)
    __shared__ __align__(16) _Float16 stage[8][16][40];

    const int tid  = threadIdx.x;
    const int w    = tid >> 5;       // wave in block
    const int lane = tid & 31;
    const int l16  = lane & 15;      // batch column n / A-row selector
    const int half = lane >> 4;      // K-half selector
    const int nrow = blockIdx.x * 128 + w * 16 + l16;   // global batch row

    // ---------------- weight tiles -> registers (A-layout f16) ----------------
    // A-layout (16-bit, 16x32): lane row M = l16; elems 0..7 -> K = half*16+i,
    // elems 8..15 -> K = half*16 + 8 + i.
    v16h whh0A[8], axb[8], wih1A[8], whh1A[8];
    v8f  biasC[8];
    #pragma unroll
    for (int g = 0; g < 8; ++g) {
        const int gate = g * 16 + l16;
        const float* r0 = W_hh0 + gate * HID;
        const float* r1 = W_ih1 + gate * HID;
        const float* r2 = W_hh1 + gate * HID;
        v16h a0, a1, a2, ax;
        #pragma unroll
        for (int i = 0; i < 8; ++i) {
            const int kA = half * 16 + i;
            const int kB = half * 16 + 8 + i;
            a0[i] = (_Float16)r0[kA]; a0[8 + i] = (_Float16)r0[kB];
            a1[i] = (_Float16)r1[kA]; a1[8 + i] = (_Float16)r1[kB];
            a2[i] = (_Float16)r2[kA]; a2[8 + i] = (_Float16)r2[kB];
        }
        whh0A[g] = a0; wih1A[g] = a1; whh1A[g] = a2;

        // rank-2 input tile: K=0 -> W_ih0 column, K=1 -> layer0 bias (half==0 lanes hold K<8)
        #pragma unroll
        for (int i = 0; i < 16; ++i) ax[i] = (_Float16)0.0f;
        ax[0] = half ? (_Float16)0.0f : (_Float16)W_ih0[gate];
        ax[1] = half ? (_Float16)0.0f : (_Float16)(b_ih0[gate] + b_hh0[gate]);
        axb[g] = ax;

        // layer1 bias as C-init: C elem r <-> gate row M = half*8 + r of tile g
        v8f bc;
        #pragma unroll
        for (int r = 0; r < 8; ++r) {
            const int m = g * 16 + half * 8 + r;
            bc[r] = b_ih1[m] + b_hh1[m];
        }
        biasC[g] = bc;
    }

    // ---------------- state ----------------
    v16h h1B, h2B;                 // B-layout: elem i <-> K = half*16 + i, col n = l16
    v8f  c1[2], c2[2];             // D-layout: elem r <-> hidden j = ch*16 + half*8 + r
    #pragma unroll
    for (int i = 0; i < 16; ++i) { h1B[i] = (_Float16)0.0f; h2B[i] = (_Float16)0.0f; }
    #pragma unroll
    for (int r = 0; r < 8; ++r) { c1[0][r] = 0.f; c1[1][r] = 0.f; c2[0][r] = 0.f; c2[1][r] = 0.f; }

    // persistent [x ; 1 ; 0...] B tile: only elem 0 changes per step
    v16h bx;
    #pragma unroll
    for (int i = 0; i < 16; ++i) bx[i] = (_Float16)0.0f;
    bx[1] = half ? (_Float16)0.0f : (_Float16)1.0f;

    const float* xp = x + (size_t)nrow * TSTEPS;   // x is [B, T, 1]
    float xv = xp[0];

    #pragma unroll 1
    for (int t = 0; t < TSTEPS; ++t) {
        const float xnext = xp[(t + 1) & (TSTEPS - 1)];   // wrap-prefetch (last value unused)

        bx[0] = half ? (_Float16)0.0f : (_Float16)xv;

        // ================= layer 1 =================
        #pragma unroll
        for (int ch = 0; ch < 2; ++ch) {
            v8f z0;
            #pragma unroll
            for (int r = 0; r < 8; ++r) z0[r] = 0.f;
            v8f zi = wmma_f16(axb[ch],     bx, z0);  zi = wmma_f16(whh0A[ch],     h1B, zi);
            v8f zf = wmma_f16(axb[2 + ch], bx, z0);  zf = wmma_f16(whh0A[2 + ch], h1B, zf);
            v8f zg = wmma_f16(axb[4 + ch], bx, z0);  zg = wmma_f16(whh0A[4 + ch], h1B, zg);
            v8f zo = wmma_f16(axb[6 + ch], bx, z0);  zo = wmma_f16(whh0A[6 + ch], h1B, zo);
            v8h hp;
            #pragma unroll
            for (int r = 0; r < 8; ++r) {
                float iv = sigm(zi[r]);
                float fv = sigm(zf[r]);
                float gv = tanh_hw(zg[r]);
                float ov = sigm(zo[r]);
                float cv = fmaf(fv, c1[ch][r], iv * gv);
                c1[ch][r] = cv;
                hp[r] = (_Float16)(ov * tanh_hw(cv));
            }
            // D->LDS: stage[n][j], j = ch*16 + half*8 + r (8 contiguous halfs, 16B aligned)
            *(v8h*)&stage[w][l16][ch * 16 + half * 8] = hp;
        }
        asm volatile("s_wait_dscnt 0x0" ::: "memory");
        h1B = *(const v16h*)&stage[w][l16][half * 16];   // B-layout reload (2x ds_load_b128)

        // ================= layer 2 =================
        #pragma unroll
        for (int ch = 0; ch < 2; ++ch) {
            v8f zi = wmma_f16(wih1A[ch],     h1B, biasC[ch]);      zi = wmma_f16(whh1A[ch],     h2B, zi);
            v8f zf = wmma_f16(wih1A[2 + ch], h1B, biasC[2 + ch]);  zf = wmma_f16(whh1A[2 + ch], h2B, zf);
            v8f zg = wmma_f16(wih1A[4 + ch], h1B, biasC[4 + ch]);  zg = wmma_f16(whh1A[4 + ch], h2B, zg);
            v8f zo = wmma_f16(wih1A[6 + ch], h1B, biasC[6 + ch]);  zo = wmma_f16(whh1A[6 + ch], h2B, zo);
            v8h hp;
            #pragma unroll
            for (int r = 0; r < 8; ++r) {
                float iv = sigm(zi[r]);
                float fv = sigm(zf[r]);
                float gv = tanh_hw(zg[r]);
                float ov = sigm(zo[r]);
                float cv = fmaf(fv, c2[ch][r], iv * gv);
                c2[ch][r] = cv;
                hp[r] = (_Float16)(ov * tanh_hw(cv));
            }
            *(v8h*)&stage[w][l16][ch * 16 + half * 8] = hp;
        }
        asm volatile("s_wait_dscnt 0x0" ::: "memory");
        h2B = *(const v16h*)&stage[w][l16][half * 16];

        xv = xnext;
    }

    // ---------------- final FC: out[b] = h2[b,:] . W_fc + b_fc ----------------
    // stage currently holds the final h2 (f16) per batch row.
    if (half == 0) {
        float acc = b_fc[0];
        #pragma unroll
        for (int j = 0; j < HID; ++j)
            acc += (float)stage[w][l16][j] * W_fc[j];
        out[nrow] = acc;
    }
}

extern "C" void kernel_launch(void* const* d_in, const int* in_sizes, int n_in,
                              void* d_out, int out_size, void* d_ws, size_t ws_size,
                              hipStream_t stream) {
    const float* x     = (const float*)d_in[0];
    const float* W_ih0 = (const float*)d_in[1];
    const float* W_hh0 = (const float*)d_in[2];
    const float* b_ih0 = (const float*)d_in[3];
    const float* b_hh0 = (const float*)d_in[4];
    const float* W_ih1 = (const float*)d_in[5];
    const float* W_hh1 = (const float*)d_in[6];
    const float* b_ih1 = (const float*)d_in[7];
    const float* b_hh1 = (const float*)d_in[8];
    const float* W_fc  = (const float*)d_in[9];
    const float* b_fc  = (const float*)d_in[10];
    float* out = (float*)d_out;

    const int Bn = in_sizes[0] / TSTEPS;      // 4096 batch rows (D=1)
    dim3 grid(Bn / 128);                      // 128 rows per block (8 waves x 16)
    lstm_wmma_kernel<<<grid, 256, 0, stream>>>(x, W_ih0, W_hh0, b_ih0, b_hh0,
                                               W_ih1, W_hh1, b_ih1, b_hh1,
                                               W_fc, b_fc, out);
}
